// TransductiveGAT_19980187861406
// MI455X (gfx1250) — matrix operand order repaired and checked by reading, not compiled, not run on hardware
//
#include <hip/hip_runtime.h>
#include <math.h>

#define N_NODES 50000
#define N_EDGES 400000
#define EE (N_EDGES + N_NODES)

typedef _Float16 half_t;
typedef __attribute__((ext_vector_type(16))) _Float16 v16h;
typedef __attribute__((ext_vector_type(8)))  float    v8f;

union Frag { v16h h; uint4 u[2]; };

// ---------- helpers ----------
__device__ __forceinline__ unsigned fenc(float f) {
    unsigned b = __float_as_uint(f);
    return (b & 0x80000000u) ? ~b : (b | 0x80000000u);
}
__device__ __forceinline__ float fdec(unsigned u) {
    unsigned b = (u & 0x80000000u) ? (u & 0x7FFFFFFFu) : ~u;
    return __uint_as_float(b);
}
__device__ __forceinline__ void edge_pair(const int* __restrict__ ei, int e, int& src, int& dst) {
    if (e < N_EDGES) { src = ei[e]; dst = ei[N_EDGES + e]; }
    else             { src = e - N_EDGES; dst = src; }
}

// ---------- small utility kernels ----------
__global__ void k_f32_to_f16(const float* __restrict__ in, half_t* __restrict__ out, int n) {
    int i = blockIdx.x * blockDim.x + threadIdx.x;
    if (i < n) out[i] = (half_t)in[i];
}

__global__ void k_transpose_to_f16(const float* __restrict__ W, half_t* __restrict__ Wt, int K, int N) {
    int i = blockIdx.x * blockDim.x + threadIdx.x;
    if (i < K * N) {
        int k = i / N, n = i % N;
        Wt[n * K + k] = (half_t)W[i];
    }
}

__global__ void k_zero_i32(int* __restrict__ p, int n) {
    int i = blockIdx.x * blockDim.x + threadIdx.x;
    if (i < n) p[i] = 0;
}

__global__ void k_bias_cat(const float* __restrict__ b, float* __restrict__ dst, int halfN) {
    int i = blockIdx.x * blockDim.x + threadIdx.x;
    if (i < 2 * halfN) dst[i] = (i < halfN) ? 0.0f : b[i - halfN];
}

__global__ void k_init_attn(unsigned* __restrict__ mx, float* __restrict__ den, int n) {
    int i = blockIdx.x * blockDim.x + threadIdx.x;
    if (i < n) { mx[i] = 0x007FFFFFu; /* fenc(-inf) */ den[i] = 0.0f; }
}

// ---------- CSR build over dst (once per call; shared by both GAT blocks) ----------
__global__ void k_count_dst(const int* __restrict__ ei, int* __restrict__ cnt) {
    int e = blockIdx.x * blockDim.x + threadIdx.x;
    if (e >= EE) return;
    int src, dst; edge_pair(ei, e, src, dst);
    atomicAdd(&cnt[dst], 1);
}

__global__ void __launch_bounds__(256)
k_scan(const int* __restrict__ cnt, int* __restrict__ rowstart) {
    __shared__ int part[256];
    const int t   = threadIdx.x;
    const int per = (N_NODES + 255) / 256;
    const int s0  = t * per;
    int s1 = s0 + per; if (s1 > N_NODES) s1 = N_NODES;
    int s = 0;
    for (int i = s0; i < s1; ++i) s += cnt[i];
    part[t] = s;
    __syncthreads();
    if (t == 0) {
        int acc = 0;
        for (int i = 0; i < 256; ++i) { int v = part[i]; part[i] = acc; acc += v; }
    }
    __syncthreads();
    int acc = part[t];
    for (int i = s0; i < s1; ++i) { rowstart[i] = acc; acc += cnt[i]; }
    if (t == 0) rowstart[N_NODES] = EE;
}

__global__ void k_fill_pos(const int* __restrict__ ei, const int* __restrict__ rowstart,
                           int* __restrict__ fill, int* __restrict__ edgeid) {
    int e = blockIdx.x * blockDim.x + threadIdx.x;
    if (e >= EE) return;
    int src, dst; edge_pair(ei, e, src, dst);
    int p = rowstart[dst] + atomicAdd(&fill[dst], 1);
    edgeid[p] = e;
}

// ---------- WMMA GEMM: C[M,N] = A[M,K](f16) * Bt[N,K]^T(f16), f32 accum ----------
// 256 threads = 8 waves; block stages one 16-row A tile in padded LDS;
// each wave owns N/16/8 output tiles. ACT: 0 = identity, 1 = leaky(0.1)
template<int ACT>
__global__ void __launch_bounds__(256)
k_gemm(const half_t* __restrict__ A, const half_t* __restrict__ Bt,
       const float* __restrict__ bias, float* __restrict__ Cf, half_t* __restrict__ Ch,
       int N, int K)
{
    extern __shared__ half_t smemA[];             // [16][K+8]
    const int tid    = threadIdx.x;
    const int m0     = blockIdx.x * 16;
    const int stride = K + 8;

    const int chunksPerRow = K >> 3;
    const int chunks = 16 * chunksPerRow;
    for (int idx = tid; idx < chunks; idx += 256) {
        int row = idx / chunksPerRow;
        int cc  = idx % chunksPerRow;
        *reinterpret_cast<uint4*>(&smemA[row * stride + cc * 8]) =
            *reinterpret_cast<const uint4*>(&A[(size_t)(m0 + row) * K + cc * 8]);
    }
    __syncthreads();

    const int wave = tid >> 5;
    const int lane = tid & 31;
    const int l    = lane & 15;
    const int hf   = lane >> 4;
    const int ntiles = N >> 4;

    for (int nt = wave; nt < ntiles; nt += 8) {
        const int n0 = nt << 4;
        v8f acc = {0.f, 0.f, 0.f, 0.f, 0.f, 0.f, 0.f, 0.f};
        for (int k0 = 0; k0 < K; k0 += 32) {
            Frag a, b;
            a.u[0] = *reinterpret_cast<const uint4*>(&smemA[l * stride + k0 + hf * 8]);
            a.u[1] = *reinterpret_cast<const uint4*>(&smemA[l * stride + k0 + 16 + hf * 8]);
            const half_t* bp = &Bt[(size_t)(n0 + l) * K + k0 + hf * 8];
            b.u[0] = *reinterpret_cast<const uint4*>(bp);
            b.u[1] = *reinterpret_cast<const uint4*>(bp + 16);
            acc = __builtin_amdgcn_wmma_f32_16x16x32_f16(false, a.h, false, b.h,
                                                         (short)0, acc, false, false);
        }
        const int col = n0 + l;
        const float bv = bias ? bias[col] : 0.0f;
#pragma unroll
        for (int r = 0; r < 8; ++r) {
            int row = m0 + hf * 8 + r;
            float v = acc[r] + bv;
            if (ACT == 1) v = v > 0.0f ? v : 0.1f * v;
            size_t o = (size_t)row * N + col;
            if (Cf) Cf[o] = v;
            if (Ch) Ch[o] = (half_t)v;
        }
    }
}

// ---------- LayerNorm(128) + ReLU, write f16 ----------
__global__ void __launch_bounds__(128)
k_ln_relu(const float* __restrict__ in, const float* __restrict__ g,
          const float* __restrict__ b, half_t* __restrict__ out)
{
    __shared__ float red[128];
    const int row = blockIdx.x, t = threadIdx.x;
    float v = in[row * 128 + t];
    red[t] = v; __syncthreads();
    for (int s = 64; s > 0; s >>= 1) { if (t < s) red[t] += red[t + s]; __syncthreads(); }
    float mu = red[0] * (1.0f / 128.0f);
    __syncthreads();
    float d = v - mu;
    red[t] = d * d; __syncthreads();
    for (int s = 64; s > 0; s >>= 1) { if (t < s) red[t] += red[t + s]; __syncthreads(); }
    float var = red[0] * (1.0f / 128.0f);
    float y = d * rsqrtf(var + 1e-5f) * g[t] + b[t];
    y = fmaxf(y, 0.0f);
    out[row * 128 + t] = (half_t)y;
}

// ---------- attention scalar scores a_s[n,h], a_d[n,h]; g is first half of GS row ----------
__global__ void k_attn_scores(const float* __restrict__ GS, int ldg,
                              const float* __restrict__ as_, const float* __restrict__ ad_,
                              float* __restrict__ a_s, float* __restrict__ a_d, int C)
{
    int i = blockIdx.x * blockDim.x + threadIdx.x;
    if (i >= N_NODES * 2) return;
    int n = i >> 1, h = i & 1;
    const float* gp  = GS + (size_t)n * ldg + h * C;
    const float* asp = as_ + h * C;
    const float* adp = ad_ + h * C;
    float ss = 0.f, sd = 0.f;
    for (int c = 0; c < C; ++c) { float gv = gp[c]; ss += gv * asp[c]; sd += gv * adp[c]; }
    a_s[i] = ss; a_d[i] = sd;
}

// ---------- edge pass 1: leaky score + segment max over dst ----------
__global__ void k_edge_max(const int* __restrict__ ei, const float* __restrict__ a_s,
                           const float* __restrict__ a_d, float* __restrict__ ev,
                           unsigned* __restrict__ mx)
{
    int i = blockIdx.x * blockDim.x + threadIdx.x;
    if (i >= EE * 2) return;
    int e = i >> 1, h = i & 1, src, dst;
    edge_pair(ei, e, src, dst);
    float v = a_s[src * 2 + h] + a_d[dst * 2 + h];
    v = v > 0.0f ? v : 0.2f * v;
    ev[i] = v;
    atomicMax(&mx[dst * 2 + h], fenc(v));
}

// ---------- edge pass 2: exp(e - max), segment sum ----------
__global__ void k_edge_exp(const int* __restrict__ ei, float* __restrict__ ev,
                           const unsigned* __restrict__ mx, float* __restrict__ den)
{
    int i = blockIdx.x * blockDim.x + threadIdx.x;
    if (i >= EE * 2) return;
    int e = i >> 1, h = i & 1, src, dst;
    edge_pair(ei, e, src, dst);
    float x = __expf(ev[i] - fdec(mx[dst * 2 + h]));
    ev[i] = x;
    atomicAdd(&den[dst * 2 + h], x);
}

// ---------- gather-aggregate + bias + skip + ELU(0.1) + f16 out (no atomics) ----------
// GS row: [0..HC) = g (2 heads x C), [HC..2HC) = skip (bias already applied in GEMM)
__global__ void __launch_bounds__(256)
k_aggregate_fuse(const int* __restrict__ rowstart, const int* __restrict__ edgeid,
                 const int* __restrict__ ei, const float* __restrict__ ev,
                 const float* __restrict__ den, const float* __restrict__ GS, int ldg,
                 const float* __restrict__ bg, half_t* __restrict__ outh, int C)
{
    __shared__ int   s_src[64];
    __shared__ float s_a0[64], s_a1[64];
    const int n  = blockIdx.x;
    const int t  = threadIdx.x;
    const int HC = 2 * C;
    const int beg = rowstart[n], end = rowstart[n + 1];
    const float d0 = 1.0f / den[n * 2 + 0];
    const float d1 = 1.0f / den[n * 2 + 1];
    float acc0 = 0.f, acc1 = 0.f;                 // channels t and (t+256 when HC==512)
    for (int cb = beg; cb < end; cb += 64) {
        int m = end - cb; if (m > 64) m = 64;
        if (t < m) {
            int e = edgeid[cb + t], src, dst;
            edge_pair(ei, e, src, dst);
            s_src[t] = src;
            s_a0[t] = ev[e * 2 + 0] * d0;
            s_a1[t] = ev[e * 2 + 1] * d1;
        }
        __syncthreads();
        for (int j = 0; j < m; ++j) {
            const float* gp = GS + (size_t)s_src[j] * ldg;
            acc0 += ((t >= C) ? s_a1[j] : s_a0[j]) * gp[t];
            if (HC == 512) acc1 += s_a1[j] * gp[t + 256];   // cols 256..511 are head 1
        }
        __syncthreads();
    }
    {
        float v = acc0 + bg[t] + GS[(size_t)n * ldg + HC + t];
        v = v > 0.0f ? v : 0.1f * expm1f(v);
        outh[(size_t)n * HC + t] = (half_t)v;
    }
    if (HC == 512) {
        int c = t + 256;
        float v = acc1 + bg[c] + GS[(size_t)n * ldg + HC + c];
        v = v > 0.0f ? v : 0.1f * expm1f(v);
        outh[(size_t)n * HC + c] = (half_t)v;
    }
}

// ================= host side =================
extern "C" void kernel_launch(void* const* d_in, const int* in_sizes, int n_in,
                              void* d_out, int out_size, void* d_ws, size_t ws_size,
                              hipStream_t stream)
{
    const float* x   = (const float*)d_in[0];
    const int*   ei  = (const int*)d_in[1];
    const float* We1 = (const float*)d_in[2];  const float* be1 = (const float*)d_in[3];
    const float* lng = (const float*)d_in[4];  const float* lnb = (const float*)d_in[5];
    const float* We2 = (const float*)d_in[6];  const float* be2 = (const float*)d_in[7];
    const float* Wg1 = (const float*)d_in[8];  const float* as1 = (const float*)d_in[9];
    const float* ad1 = (const float*)d_in[10]; const float* bg1 = (const float*)d_in[11];
    const float* Ws1 = (const float*)d_in[12]; const float* bs1 = (const float*)d_in[13];
    const float* Wg2 = (const float*)d_in[14]; const float* as2 = (const float*)d_in[15];
    const float* ad2 = (const float*)d_in[16]; const float* bg2 = (const float*)d_in[17];
    const float* Ws2 = (const float*)d_in[18]; const float* bs2 = (const float*)d_in[19];
    const float* Wd1 = (const float*)d_in[20]; const float* bd1 = (const float*)d_in[21];
    const float* Wd2 = (const float*)d_in[22]; const float* bd2 = (const float*)d_in[23];
    float* out = (float*)d_out;

    char* base = (char*)d_ws;
    size_t off = 0;
    auto alloc = [&](size_t bytes) -> void* {
        void* p = (void*)(base + off);
        off += (bytes + 255) & ~(size_t)255;
        return p;
    };

    float*    GS   = (float*)alloc((size_t)N_NODES * 1024 * 4);   // g || skip (block1: ld 1024)
    float*    F0   = (float*)alloc((size_t)N_NODES * 128 * 4);    // encoder pre-LN
    half_t*   HBF  = (half_t*)alloc((size_t)N_NODES * 512 * 2);   // current features (f16)
    half_t*   ABF  = (half_t*)alloc((size_t)N_NODES * 512 * 2);   // alt f16 buffer
    float*    EV   = (float*)alloc((size_t)EE * 2 * 4);
    float*    ASC  = (float*)alloc((size_t)N_NODES * 2 * 4);
    float*    ADC  = (float*)alloc((size_t)N_NODES * 2 * 4);
    unsigned* MX   = (unsigned*)alloc((size_t)N_NODES * 2 * 4);
    float*    DEN  = (float*)alloc((size_t)N_NODES * 2 * 4);
    int*      CNT  = (int*)alloc((size_t)N_NODES * 4);            // counts, then fill cursors
    int*      ROWS = (int*)alloc((size_t)(N_NODES + 1) * 4);
    int*      EID  = (int*)alloc((size_t)EE * 4);
    half_t* We1t  = (half_t*)alloc(128 * 128 * 2);
    half_t* We2t  = (half_t*)alloc(128 * 128 * 2);
    half_t* Wcat1 = (half_t*)alloc(1024 * 128 * 2);               // [Wg1t ; Ws1t]
    half_t* Wcat2 = (half_t*)alloc(512 * 512 * 2);                // [Wg2t ; Ws2t]
    half_t* Wd1t  = (half_t*)alloc(256 * 128 * 2);
    half_t* Wd2t  = (half_t*)alloc(128 * 128 * 2);
    float*  BC1   = (float*)alloc(1024 * 4);                      // 0 || bs1
    float*  BC2   = (float*)alloc(512 * 4);                       // 0 || bs2

    const int TPB = 256;
    auto nb = [](int n) { return (n + 255) / 256; };

    // ---- weight prep ----
    k_transpose_to_f16<<<nb(128 * 128), TPB, 0, stream>>>(We1, We1t, 128, 128);
    k_transpose_to_f16<<<nb(128 * 128), TPB, 0, stream>>>(We2, We2t, 128, 128);
    k_transpose_to_f16<<<nb(128 * 512), TPB, 0, stream>>>(Wg1, Wcat1, 128, 512);
    k_transpose_to_f16<<<nb(128 * 512), TPB, 0, stream>>>(Ws1, Wcat1 + 512 * 128, 128, 512);
    k_transpose_to_f16<<<nb(512 * 256), TPB, 0, stream>>>(Wg2, Wcat2, 512, 256);
    k_transpose_to_f16<<<nb(512 * 256), TPB, 0, stream>>>(Ws2, Wcat2 + 256 * 512, 512, 256);
    k_transpose_to_f16<<<nb(256 * 128), TPB, 0, stream>>>(Wd1, Wd1t, 256, 128);
    k_transpose_to_f16<<<nb(128 * 128), TPB, 0, stream>>>(Wd2, Wd2t, 128, 128);
    k_bias_cat<<<nb(1024), TPB, 0, stream>>>(bs1, BC1, 512);
    k_bias_cat<<<nb(512), TPB, 0, stream>>>(bs2, BC2, 256);

    // ---- CSR over dst (shared by both GAT blocks) ----
    k_zero_i32<<<nb(N_NODES), TPB, 0, stream>>>(CNT, N_NODES);
    k_count_dst<<<nb(EE), TPB, 0, stream>>>(ei, CNT);
    k_scan<<<1, 256, 0, stream>>>(CNT, ROWS);
    k_zero_i32<<<nb(N_NODES), TPB, 0, stream>>>(CNT, N_NODES);
    k_fill_pos<<<nb(EE), TPB, 0, stream>>>(ei, ROWS, CNT, EID);

    const dim3 ggrid(N_NODES / 16);   // 3125 exact M-tiles
    const size_t smem128 = 16 * (128 + 8) * sizeof(half_t);
    const size_t smem256 = 16 * (256 + 8) * sizeof(half_t);
    const size_t smem512 = 16 * (512 + 8) * sizeof(half_t);

    // ---- encoder ----
    k_f32_to_f16<<<nb(N_NODES * 128), TPB, 0, stream>>>(x, ABF, N_NODES * 128);
    k_gemm<0><<<ggrid, TPB, smem128, stream>>>(ABF, We1t, be1, F0, nullptr, 128, 128);
    k_ln_relu<<<N_NODES, 128, 0, stream>>>(F0, lng, lnb, ABF);
    k_gemm<0><<<ggrid, TPB, smem128, stream>>>(ABF, We2t, be2, nullptr, HBF, 128, 128);

    // ---- GAT block 1 (128 -> 2x256, fused g||skip GEMM, ld 1024) ----
    k_gemm<0><<<ggrid, TPB, smem128, stream>>>(HBF, Wcat1, BC1, GS, nullptr, 1024, 128);
    k_attn_scores<<<nb(N_NODES * 2), TPB, 0, stream>>>(GS, 1024, as1, ad1, ASC, ADC, 256);
    k_init_attn<<<nb(N_NODES * 2), TPB, 0, stream>>>(MX, DEN, N_NODES * 2);
    k_edge_max<<<nb(EE * 2), TPB, 0, stream>>>(ei, ASC, ADC, EV, MX);
    k_edge_exp<<<nb(EE * 2), TPB, 0, stream>>>(ei, EV, MX, DEN);
    k_aggregate_fuse<<<N_NODES, TPB, 0, stream>>>(ROWS, EID, ei, EV, DEN, GS, 1024, bg1, HBF, 256);

    // ---- GAT block 2 (512 -> 2x128, fused g||skip GEMM, ld 512) ----
    k_gemm<0><<<ggrid, TPB, smem512, stream>>>(HBF, Wcat2, BC2, GS, nullptr, 512, 512);
    k_attn_scores<<<nb(N_NODES * 2), TPB, 0, stream>>>(GS, 512, as2, ad2, ASC, ADC, 128);
    k_init_attn<<<nb(N_NODES * 2), TPB, 0, stream>>>(MX, DEN, N_NODES * 2);
    k_edge_max<<<nb(EE * 2), TPB, 0, stream>>>(ei, ASC, ADC, EV, MX);
    k_edge_exp<<<nb(EE * 2), TPB, 0, stream>>>(ei, EV, MX, DEN);
    k_aggregate_fuse<<<N_NODES, TPB, 0, stream>>>(ROWS, EID, ei, EV, DEN, GS, 512, bg2, HBF, 128);

    // ---- decoder ----
    k_gemm<1><<<ggrid, TPB, smem256, stream>>>(HBF, Wd1t, bd1, nullptr, ABF, 128, 256);
    k_gemm<0><<<ggrid, TPB, smem128, stream>>>(ABF, Wd2t, bd2, out, nullptr, 128, 128);
}
